// Recon_35201551958416
// MI455X (gfx1250) — matrix operand order repaired and checked by reading, not compiled, not run on hardware
//
#include <hip/hip_runtime.h>
#include <hip/hip_bf16.h>
#include <stdint.h>

// ---------------------------------------------------------------------------
// Types for CDNA5 WMMA (wave32): v_wmma_f32_16x16x32_bf16
// ---------------------------------------------------------------------------
typedef __attribute__((ext_vector_type(16))) __bf16 v16bf;
typedef __attribute__((ext_vector_type(8)))  float  v8f;

// TDM descriptor group types (probe-verified builtin signature)
typedef __attribute__((ext_vector_type(4))) unsigned int tdm_g0_t;  // uint32x4
typedef __attribute__((ext_vector_type(8))) int          tdm_g1_t;  // int32x8
typedef __attribute__((ext_vector_type(4))) int          tdm_g2_t;  // int32x4

#if __has_include(<hip/amd_detail/amd_gfx1250_TDM.h>)
#define TDM_6ARG 1      // therock/clang-23 toolchain: 6-arg builtin
#else
#define TDM_6ARG 0      // ROCm 7.2 / clang-22: 5-arg builtin
#endif

#define TPX 64            // output pixels per block (4 waves x 16)
#define OCT 32            // output channels per block (2 subtiles x 16)
#define CONV_THREADS 256  // 8 waves

// ---------------------------------------------------------------------------
// Issue one TDM 2D tile load (bf16 elements) into LDS.
//   tile: tile_d1 rows of tile_d0 elements, row stride stride0 (elements).
//   LDS receives rows back-to-back (linear), matching our LDS tile layouts.
// ---------------------------------------------------------------------------
__device__ __forceinline__ void tdm_load_2d(unsigned long long gaddr,
                                            unsigned int lds_off,
                                            unsigned int tile_d0,   // <= 65535
                                            unsigned int tile_d1,
                                            unsigned long long stride0_elems)
{
    tdm_g0_t g0;
    g0.x = 1u;                                            // count=1 (valid, user mode)
    g0.y = lds_off;                                       // LDS byte address
    g0.z = (unsigned int)gaddr;                           // global_addr[31:0]
    g0.w = ((unsigned int)(gaddr >> 32) & 0x01FFFFFFu)    // global_addr[56:32]
           | (2u << 30);                                  // type=2 ("image")

    tdm_g1_t g1;
    g1[0] = 0x00010000;                                   // data_size=1 -> 2 bytes
    g1[1] = (int)((tile_d0 & 0xFFFFu) << 16);             // tensor_dim0[15:0]
    g1[2] = (int)((tile_d0 >> 16) | ((tile_d1 & 0xFFFFu) << 16)); // dim0 hi | dim1 lo
    g1[3] = (int)((tile_d0 & 0xFFFFu) << 16);             // dim1 hi(=0) | tile_dim0
    g1[4] = (int)(tile_d1 & 0xFFFFu);                     // tile_dim1 | tile_dim2=0
    g1[5] = (int)(unsigned int)stride0_elems;             // tensor_dim0_stride[31:0]
    g1[6] = (int)((stride0_elems >> 32) & 0xFFFFull);     // stride[47:32] | d1stride lo=0
    g1[7] = 0;

    tdm_g2_t gz = {0, 0, 0, 0};
#if TDM_6ARG
    tdm_g1_t g4 = {0, 0, 0, 0, 0, 0, 0, 0};
    __builtin_amdgcn_tensor_load_to_lds(g0, g1, gz, gz, g4, 0);
#else
    __builtin_amdgcn_tensor_load_to_lds(g0, g1, gz, gz, 0);
#endif
}

// ---------------------------------------------------------------------------
// Pack activation NCHW f32 -> zero-halo channels-last bf16:
//   P[b][y][xx][ic] (y in 0..H+1, xx in 0..Wpitch-1, ic in 0..ICp-1)
//   = X[b][ic][y-1][xx-1]  (0 outside image / for ic >= IC)
// ---------------------------------------------------------------------------
__global__ void pack_act_kernel(const float* __restrict__ X, __bf16* __restrict__ P,
                                int Bn, int IC, int H, int W, int ICp, int Wpitch)
{
    long idx = (long)blockIdx.x * blockDim.x + threadIdx.x;
    long total = (long)Bn * (H + 2) * Wpitch * ICp;
    if (idx >= total) return;
    int ic = idx % ICp;
    long t = idx / ICp;
    int xx = t % Wpitch;
    t /= Wpitch;
    int y = t % (H + 2);
    int b = t / (H + 2);
    int gy = y - 1, gx = xx - 1;
    float v = 0.0f;
    if (ic < IC && gy >= 0 && gy < H && gx >= 0 && gx < W)
        v = X[(((size_t)b * IC + ic) * H + gy) * W + gx];
    P[idx] = (__bf16)v;
}

// ---------------------------------------------------------------------------
// Pack weights OIHW f32 -> per-octile contiguous bf16 slabs:
//   Pw[t][m(0..OCT-1)][kk(0..8)][ic(0..ICp-1)],  oc = t*OCT + m (0-padded)
// ---------------------------------------------------------------------------
__global__ void pack_wgt_kernel(const float* __restrict__ Wt, __bf16* __restrict__ Pw,
                                int IC, int OC, int ICp, int nOcT)
{
    int idx = blockIdx.x * blockDim.x + threadIdx.x;
    int total = nOcT * OCT * 9 * ICp;
    if (idx >= total) return;
    int ic = idx % ICp;
    int t  = idx / ICp;
    int kk = t % 9;
    t /= 9;
    int m  = t % OCT;
    int oc = (t / OCT) * OCT + m;
    float v = 0.0f;
    if (oc < OC && ic < IC) {
        int kh = kk / 3, kw = kk - kh * 3;
        v = Wt[((oc * IC + ic) * 3 + kh) * 3 + kw];
    }
    Pw[idx] = (__bf16)v;
}

// ---------------------------------------------------------------------------
// Implicit-GEMM 3x3 SAME conv. Input: packed halo bf16 [B][H+2][Wpitch][ICp].
// Weights: packed bf16 slabs. TDM stages both tiles into LDS; 8 waves do
// v_wmma_f32_16x16x32_bf16 over kh,kw,ic-chunks; fused bias + LeakyReLU(0.1);
// output f32 NCHW.
// grid = ( W/64, H, B * ceil(OC/32) ), block = 256.
// Dynamic LDS: (OCT*9*ICp + 3*66*ICp) bf16.
// ---------------------------------------------------------------------------
__global__ __launch_bounds__(CONV_THREADS, 1)
void conv3x3_wmma_tdm(const __bf16* __restrict__ Xp, const __bf16* __restrict__ Wp,
                      const float* __restrict__ Bias, float* __restrict__ Y,
                      int Bn, int OC, int H, int W, int ICp, int Wpitch)
{
    extern __shared__ __bf16 smem[];
    __bf16* wS  = smem;                       // [OCT][9*ICp]
    __bf16* inS = smem + OCT * 9 * ICp;       // [3][66][ICp]

    const int tid  = threadIdx.x;
    const int lane = tid & 31;
    const int wave = tid >> 5;

    const int nOcT  = (OC + OCT - 1) / OCT;
    const int b     = blockIdx.z / nOcT;
    const int ocT   = blockIdx.z % nOcT;
    const int ocBlk = ocT * OCT;
    const int yrow  = blockIdx.y;
    const int x0    = blockIdx.x * TPX;

    // ---- TDM: DMA weight slab + input halo tile into LDS (wave 0 only) ----
    if (wave == 0) {
        const unsigned int wElems = (unsigned int)(OCT * 9 * ICp);
        unsigned long long wg = (unsigned long long)(uintptr_t)Wp
                              + (unsigned long long)ocT * wElems * 2ull;
        tdm_load_2d(wg, (unsigned int)(uintptr_t)wS, wElems, 1u, 0ull);

        unsigned long long ig = (unsigned long long)(uintptr_t)Xp
            + ((((unsigned long long)b * (H + 2) + yrow) * Wpitch + x0)
               * (unsigned long long)ICp) * 2ull;
        tdm_load_2d(ig, (unsigned int)(uintptr_t)inS,
                    (unsigned int)(66 * ICp), 3u,
                    (unsigned long long)Wpitch * (unsigned long long)ICp);

        __builtin_amdgcn_s_wait_tensorcnt(0);
    }
    __syncthreads();

    // wave -> (oc subtile, pixel subtile)
    const int ocSub  = wave >> 2;               // 0..1
    const int pxw    = (wave & 3) * 16;         // 0,16,32,48
    const int ocBase = ocBlk + ocSub * 16;

    const int nrow = lane & 15;                 // A: M row / B: N column
    const int half = lane >> 4;                 // K half selector

    v8f acc = {0.f, 0.f, 0.f, 0.f, 0.f, 0.f, 0.f, 0.f};

    for (int kk = 0; kk < 9; ++kk) {
        const int kh = kk / 3;
        const int kw = kk - kh * 3;
        const __bf16* aRow = wS + (ocSub * 16 + nrow) * (9 * ICp) + kk * ICp;
        const __bf16* bRow = inS + ((kh * 66) + (pxw + nrow + kw)) * ICp;
        for (int kc = 0; kc < ICp; kc += 32) {
            // A fragment: 16x32 bf16; lane halves hold K = half*8+{0..7} and
            // K = 16+half*8+{0..7}  -> two 16B LDS loads.
            union { uint4 u[2]; v16bf v; } A;
            const __bf16* ap = aRow + kc + half * 8;
            A.u[0] = *(const uint4*)(ap);
            A.u[1] = *(const uint4*)(ap + 16);
            // B fragment: 32x16 bf16; lane halves hold K = half*8+{0..7} and
            // K = 16+half*8+{0..7}, column N = nrow  -> two 16B LDS loads.
            union { uint4 u[2]; v16bf v; } Bf;
            const __bf16* bp = bRow + kc + half * 8;
            Bf.u[0] = *(const uint4*)(bp);
            Bf.u[1] = *(const uint4*)(bp + 16);

            acc = __builtin_amdgcn_wmma_f32_16x16x32_bf16(
                false, A.v, false, Bf.v, (short)0, acc, false, false);
        }
    }

    // ---- store C tile: VGPR v holds (M = v + 8*half, N = nrow) ----
    const int px = x0 + pxw + nrow;
    if (px < W) {
        #pragma unroll
        for (int v = 0; v < 8; ++v) {
            int oc = ocBase + v + half * 8;
            if (oc < OC) {
                float r = acc[v] + Bias[oc];
                r = (r >= 0.0f) ? r : 0.1f * r;
                Y[(((size_t)b * OC + oc) * H + yrow) * W + px] = r;
            }
        }
    }
}

// ---------------------------------------------------------------------------
// Fused 2D fractional max pool (PyTorch-style intervals, k=4, u=0.5).
// ---------------------------------------------------------------------------
__device__ __forceinline__ int fmp_start(int i, int n, int out)
{
    if (i >= out - 1) return n - 4;
    float alpha = (float)(n - 4) / (float)(out - 1);
    int s = (int)(floorf(((float)i + 0.5f) * alpha) - floorf(0.5f * alpha));
    if (s < 0) s = 0;
    if (s > n - 4) s = n - 4;
    return s;
}

__global__ void fmp2d_kernel(const float* __restrict__ in, float* __restrict__ out,
                             int BC, int Hin, int Win, int Ho, int Wo)
{
    int idx = blockIdx.x * blockDim.x + threadIdx.x;
    int total = BC * Ho * Wo;
    if (idx >= total) return;
    int xo = idx % Wo;
    int t  = idx / Wo;
    int yo = t % Ho;
    int bc = t / Ho;
    int sy = fmp_start(yo, Hin, Ho);
    int sx = fmp_start(xo, Win, Wo);
    const float* base = in + ((size_t)bc * Hin + sy) * Win + sx;
    float m = -3.4e38f;
    #pragma unroll
    for (int dy = 0; dy < 4; ++dy)
        #pragma unroll
        for (int dx = 0; dx < 4; ++dx) {
            float v = base[dy * Win + dx];
            m = (v > m) ? v : m;
        }
    out[idx] = m;
}

// ---------------------------------------------------------------------------
// out[b, 0..C1+C2, y, x] = concat( up2(up_src)[C1 @ Hs,Ws], cat_src[C2 @ H,W] )
// ---------------------------------------------------------------------------
__global__ void up2cat_kernel(const float* __restrict__ up_src, int C1, int Hs, int Ws,
                              const float* __restrict__ cat_src, int C2,
                              float* __restrict__ out, int Bn, int H, int W)
{
    long idx = (long)blockIdx.x * blockDim.x + threadIdx.x;
    int Ct = C1 + C2;
    long total = (long)Bn * Ct * H * W;
    if (idx >= total) return;
    int x = idx % W;
    long t = idx / W;
    int y = t % H;
    t /= H;
    int c = t % Ct;
    int b = t / Ct;
    float v;
    if (c < C1)
        v = up_src[(((size_t)b * C1 + c) * Hs + (y >> 1)) * Ws + (x >> 1)];
    else
        v = cat_src[(((size_t)b * C2 + (c - C1)) * H + y) * W + x];
    out[idx] = v;
}

// ---------------------------------------------------------------------------
// xt2[b,t,h,x] = Cu[t,h,x]/28 * sum_i Cu[i,h,2t+x-2i] * xt[b,i,h,2t+x-2i]
// out = (1 - deta*eta)*xt - deta*xt2 + deta*x + deta*eta*z
// ---------------------------------------------------------------------------
__global__ void combine_kernel(const float* __restrict__ xt, const float* __restrict__ x,
                               const float* __restrict__ Cu,
                               const float* __restrict__ deta_p, const float* __restrict__ eta_p,
                               const float* __restrict__ z, float* __restrict__ out,
                               int Bn, int C, int H, int W)
{
    long idx = (long)blockIdx.x * blockDim.x + threadIdx.x;
    long total = (long)Bn * C * H * W;
    if (idx >= total) return;
    int xc = idx % W;
    long t = idx / W;
    int h  = t % H;
    t /= H;
    int ch = t % C;
    int b  = t / C;

    float deta = deta_p[0];
    float eta  = eta_p[0];

    int base = 2 * ch + xc;
    float acc = 0.0f;
    #pragma unroll
    for (int i = 0; i < 28; ++i) {
        int w = base - 2 * i;
        if (w >= 0 && w < W)
            acc += Cu[((size_t)i * H + h) * W + w]
                 * xt[(((size_t)b * C + i) * H + h) * W + w];
    }
    float xt2 = Cu[((size_t)ch * H + h) * W + xc] * acc * (1.0f / 28.0f);
    out[idx] = (1.0f - deta * eta) * xt[idx] - deta * xt2 + deta * x[idx]
             + deta * eta * z[idx];
}

// ---------------------------------------------------------------------------
// Host-side orchestration
// ---------------------------------------------------------------------------
extern "C" void kernel_launch(void* const* d_in, const int* in_sizes, int n_in,
                              void* d_out, int out_size, void* d_ws, size_t ws_size,
                              hipStream_t stream)
{
    const float* xt   = (const float*)d_in[0];
    const float* x    = (const float*)d_in[1];
    const float* Cu   = (const float*)d_in[2];
    const float* deta = (const float*)d_in[3];
    const float* eta  = (const float*)d_in[4];
    const float* Wk[9];
    const float* Bk[9];
    for (int j = 0; j < 9; ++j) {
        Wk[j] = (const float*)d_in[5 + 2 * j];
        Bk[j] = (const float*)d_in[6 + 2 * j];
    }

    const int Bn = 8, C = 28, H0 = 256;
    const long N = (long)Bn * C * H0 * H0;              // 14,680,064

    // Workspace: R0/R1 ping-pong (8*124*256*256 f32 each) + skip buffer +
    // packed-activation bf16 region + packed-weight bf16 region. ~690 MB.
    const size_t R = (size_t)Bn * 124 * 256 * 256;      // 65,011,712 floats
    float*  R0   = (float*)d_ws;
    float*  R1   = R0 + R;
    float*  Sk   = R1 + R;                              // 8*48*128*128 floats
    __bf16* Pact = (__bf16*)(Sk + (size_t)Bn * 48 * 128 * 128);
    __bf16* Pwgt = Pact + (size_t)Bn * 258 * 258 * 128; // max packed activation
    float*  zOut = (float*)d_out + N;
    float*  xOut = (float*)d_out;

    auto conv = [&](const float* in, float* out, int layer,
                    int IC, int OC, int H, int W) {
        int ICp = ((IC + 31) / 32) * 32;
        int ntx = (W + TPX - 1) / TPX;
        int Wpitch = ntx * TPX + 2;
        int nOcT = (OC + OCT - 1) / OCT;
        // pack activation (with zero halo) and weights to bf16
        {
            long tot = (long)Bn * (H + 2) * Wpitch * ICp;
            pack_act_kernel<<<dim3((tot + 255) / 256), dim3(256), 0, stream>>>(
                in, Pact, Bn, IC, H, W, ICp, Wpitch);
            int wt = nOcT * OCT * 9 * ICp;
            pack_wgt_kernel<<<dim3((wt + 255) / 256), dim3(256), 0, stream>>>(
                Wk[layer], Pwgt, IC, OC, ICp, nOcT);
        }
        size_t smemB = (size_t)(OCT * 9 * ICp + 3 * 66 * ICp) * sizeof(__bf16);
        dim3 grid(ntx, H, Bn * nOcT);
        conv3x3_wmma_tdm<<<grid, dim3(CONV_THREADS), smemB, stream>>>(
            Pact, Pwgt, Bk[layer], out, Bn, OC, H, W, ICp, Wpitch);
    };
    auto fmp = [&](const float* in, float* out, int Ch, int Hin, int Ho) {
        long total = (long)Bn * Ch * Ho * Ho;
        fmp2d_kernel<<<dim3((total + 255) / 256), dim3(256), 0, stream>>>(
            in, out, Bn * Ch, Hin, Hin, Ho, Ho);
    };

    // ---- UNet encoder ----
    conv(xt, R0, 0, 28, 48, 256, 256);                 // conv1
    conv(R0, R1, 1, 48, 48, 256, 256);                 // conv2
    fmp(R1, R0, 48, 256, 204);                         // fmp 0.8
    fmp(R0, Sk, 48, 204, 128);                         // fmp 0.63 -> skip
    conv(Sk, R0, 2, 48, 48, 128, 128);                 // conv3
    fmp(R0, R1, 48, 128, 102);
    fmp(R1, R0, 48, 102, 64);
    conv(R0, R1, 3, 48, 48, 64, 64);                   // conv4

    // ---- decoder ----
    {   // concat(up2(conv4), skip) -> [8,96,128,128]
        long total = (long)Bn * 96 * 128 * 128;
        up2cat_kernel<<<dim3((total + 255) / 256), dim3(256), 0, stream>>>(
            R1, 48, 64, 64, Sk, 48, R0, Bn, 128, 128);
    }
    conv(R0, R1, 4, 96, 96, 128, 128);                 // conv5
    conv(R1, R0, 5, 96, 96, 128, 128);                 // conv6
    {   // concat(up2(conv6), x0) -> [8,124,256,256]
        long total = (long)Bn * 124 * 256 * 256;
        up2cat_kernel<<<dim3((total + 255) / 256), dim3(256), 0, stream>>>(
            R0, 96, 128, 128, xt, 28, R1, Bn, 256, 256);
    }
    conv(R1, R0, 6, 124, 64, 256, 256);                // conv7
    conv(R0, R1, 7, 64, 32, 256, 256);                 // conv8
    conv(R1, zOut, 8, 32, 28, 256, 256);               // conv9 -> z

    // ---- physics term + combine -> xt_new ----
    combine_kernel<<<dim3((N + 255) / 256), dim3(256), 0, stream>>>(
        xt, x, Cu, deta, eta, zOut, xOut, Bn, C, H0, H0);
}
